// RNNfromScratch_58025008169378
// MI455X (gfx1250) — matrix operand order repaired and checked by reading, not compiled, not run on hardware
//
#include <hip/hip_runtime.h>
#include <hip/hip_bf16.h>

typedef __attribute__((ext_vector_type(16))) _Float16 v16h;
typedef __attribute__((ext_vector_type(8)))  float    v8f;

#define B_   512
#define T_   2048
#define IN_  64
#define H_   128
#define OUT_ 10
#define BTILE 16   // WMMA M per workgroup
#define NWAVES 8   // 8 waves * 16 N-cols = H

// ---------------------------------------------------------------------------
// Hardware tanh (CDNA5 TRANS op, co-executes with WMMA) instead of the
// branchy ocml tanhf expansion.
__device__ __forceinline__ float htanh(float x) {
#if __has_builtin(__builtin_amdgcn_tanhf)
    return __builtin_amdgcn_tanhf(x);
#else
    float r;
    asm("v_tanh_f32 %0, %1" : "=v"(r) : "v"(x));
    return r;
#endif
}

// ---------------------------------------------------------------------------
// LDS hidden-state buffers are stored pre-swizzled in the CDNA5 WMMA A-matrix
// fragment layout (16-bit A, 16x32 per chunk, CDNA5 ISA §7.12.2):
//   lane<16  holds row=lane,    K = {0..7, 16..23} of the chunk
//   lane>=16 holds row=lane-16, K = {8..15, 24..31} of the chunk
// so each lane's fragment is 16 contiguous f16 (32B) -> 2x ds_load_b128.
__device__ __forceinline__ int hoff(int row, int col) {
    int kk    = col & 31;
    int tlane = row + (((kk >> 3) & 1) << 4);
    int idx   = (kk & 7) + (((kk >> 4) & 1) << 3);
    return ((col >> 5) << 9) + (tlane << 4) + idx;   // chunk*512 + lane*16 + idx
}

// A-fragment: 32B contiguous per lane
__device__ __forceinline__ v16h ld_frag(const _Float16* buf, int lane, int c) {
    return *(const v16h*)(buf + (c << 9) + (lane << 4));
}

// B-matrix fragment (K x N) from a row-major [N, K] weight (W[n,k]):
//   lane<16:  n = n0+lane,    K = c*32 + {0..7, 16..23}
//   lane>=16: n = n0+lane-16, K = c*32 + {8..15, 24..31}
__device__ __forceinline__ v16h ld_wfrag(const float* W, int Kdim, int n, int c, int lane) {
    const float* p = W + (size_t)n * Kdim + c * 32 + ((lane >= 16) ? 8 : 0);
    v16h f;
#pragma unroll
    for (int i = 0; i < 16; ++i) f[i] = (_Float16)p[i + ((i >> 3) << 3)];
    return f;
}

// A-fragment for the x tile, gathered straight from global memory:
// two contiguous runs of 8 f32 matching the K-interleave above.
__device__ __forceinline__ v16h ld_xfrag(const float* xrow, int c, int hi) {
    const float* p0 = xrow + c * 32 + (hi ? 8 : 0);
    const float* p1 = p0 + 16;
    v16h f;
#pragma unroll
    for (int i = 0; i < 8; ++i) {
        f[i]     = (_Float16)p0[i];
        f[i + 8] = (_Float16)p1[i];
    }
    return f;
}

__device__ __forceinline__ v8f wmma16(v16h a, v16h b, v8f c) {
    return __builtin_amdgcn_wmma_f32_16x16x32_f16(false, a, false, b, (short)0, c,
                                                  false, false);
}

__global__ __launch_bounds__(NWAVES * 32, 1) void rnn_scan_kernel(
    const float* __restrict__ x,    const float* __restrict__ hidden,
    const float* __restrict__ Wi0,  const float* __restrict__ bi0,
    const float* __restrict__ Wh0,  const float* __restrict__ bh0,
    const float* __restrict__ Wi1,  const float* __restrict__ bi1,
    const float* __restrict__ Wh1,  const float* __restrict__ bh1,
    const float* __restrict__ Wfc,  const float* __restrict__ bfc,
    float* __restrict__ out)
{
    __shared__ __align__(32) _Float16 h0buf[2][BTILE * H_];  // double-buffered, A-layout
    __shared__ __align__(32) _Float16 h1buf[2][BTILE * H_];

    const int tid  = threadIdx.x;
    const int lane = tid & 31;
    const int wave = tid >> 5;
    const int b0   = blockIdx.x * BTILE;
    const int n0   = wave << 4;
    const int row  = lane & 15;
    const int hi   = lane >> 4;
    const int n    = n0 + row;          // this lane's output column (D layout N=lane&15)

    // --- weights -> registers, pre-swizzled into B-fragment layout (done once) ---
    v16h wi0[2], wh0[4], wi1[4], wh1[4];
#pragma unroll
    for (int c = 0; c < 2; ++c) wi0[c] = ld_wfrag(Wi0, IN_, n, c, lane);
#pragma unroll
    for (int c = 0; c < 4; ++c) wh0[c] = ld_wfrag(Wh0, H_, n, c, lane);
#pragma unroll
    for (int c = 0; c < 4; ++c) wi1[c] = ld_wfrag(Wi1, H_, n, c, lane);
#pragma unroll
    for (int c = 0; c < 4; ++c) wh1[c] = ld_wfrag(Wh1, H_, n, c, lane);
    const float bias0 = bi0[n] + bh0[n];
    const float bias1 = bi1[n] + bh1[n];

    // --- initial hidden state -> LDS (A-fragment layout) ---
    for (int j = tid; j < BTILE * H_; j += NWAVES * 32) {
        int r = j >> 7, c = j & 127;
        h0buf[0][hoff(r, c)] = (_Float16)hidden[(size_t)(b0 + r) * H_ + c];
        h1buf[0][hoff(r, c)] = (_Float16)hidden[(size_t)B_ * H_ + (size_t)(b0 + r) * H_ + c];
    }
    __syncthreads();

    const float* xrow = x + ((size_t)(b0 + row) * T_) * IN_;

    int cur = 0;
    for (int t = 0; t < T_; ++t) {
        const float* xp = xrow + (size_t)t * IN_;
        if (t + 1 < T_) __builtin_prefetch(xp + IN_, 0, 1);   // global_prefetch next step

        // Layer-1 recurrent fragments were finalized before last step's closing
        // barrier -> legal to load NOW; latency hides behind layer-0's 6 WMMAs.
        v16h h1a[4];
#pragma unroll
        for (int c = 0; c < 4; ++c) h1a[c] = ld_frag(h1buf[cur], lane, c);

        // ---- layer 0: acc = x_t @ Wi0^T + h0 @ Wh0^T ----
        v8f acc = {};
#pragma unroll
        for (int c = 0; c < 2; ++c)
            acc = wmma16(ld_xfrag(xp, c, hi), wi0[c], acc);
#pragma unroll
        for (int c = 0; c < 4; ++c)
            acc = wmma16(ld_frag(h0buf[cur], lane, c), wh0[c], acc);

        // tanh (hardware TRANS) + scatter into A-layout of the other buffer
#pragma unroll
        for (int r = 0; r < 8; ++r) {
            float v = htanh(acc[r] + bias0);
            h0buf[cur ^ 1][hoff(r + (hi << 3), n)] = (_Float16)v;
        }
        __syncthreads();   // h0_new visible to all waves

        // ---- layer 1: acc1 = h0_new @ Wi1^T + h1 @ Wh1^T ----
        v8f acc1 = {};
#pragma unroll
        for (int c = 0; c < 4; ++c)
            acc1 = wmma16(ld_frag(h0buf[cur ^ 1], lane, c), wi1[c], acc1);
#pragma unroll
        for (int c = 0; c < 4; ++c)
            acc1 = wmma16(h1a[c], wh1[c], acc1);

#pragma unroll
        for (int r = 0; r < 8; ++r) {
            float v = htanh(acc1[r] + bias1);
            h1buf[cur ^ 1][hoff(r + (hi << 3), n)] = (_Float16)v;
        }
        cur ^= 1;
        __syncthreads();   // end-of-step: both new states visible
    }

    // ---- epilogue: hidden_out (2,B,H) then output = h1 @ Wfc^T + bfc ----
    float* hidOut = out + (size_t)B_ * OUT_;
    for (int j = tid; j < BTILE * H_; j += NWAVES * 32) {
        int r = j >> 7, c = j & 127;
        hidOut[(size_t)(b0 + r) * H_ + c]                    = (float)h0buf[cur][hoff(r, c)];
        hidOut[(size_t)B_ * H_ + (size_t)(b0 + r) * H_ + c]  = (float)h1buf[cur][hoff(r, c)];
    }
    for (int j = tid; j < BTILE * OUT_; j += NWAVES * 32) {
        int r = j / OUT_, o = j % OUT_;
        float s = bfc[o];
        for (int k = 0; k < H_; ++k)
            s += (float)h1buf[cur][hoff(r, k)] * Wfc[o * H_ + k];
        out[(size_t)(b0 + r) * OUT_ + o] = s;
    }
}

extern "C" void kernel_launch(void* const* d_in, const int* in_sizes, int n_in,
                              void* d_out, int out_size, void* d_ws, size_t ws_size,
                              hipStream_t stream) {
    (void)in_sizes; (void)n_in; (void)d_ws; (void)ws_size; (void)out_size;
    const float* x   = (const float*)d_in[0];
    const float* hid = (const float*)d_in[1];
    const float* Wi0 = (const float*)d_in[2];
    const float* bi0 = (const float*)d_in[3];
    const float* Wh0 = (const float*)d_in[4];
    const float* bh0 = (const float*)d_in[5];
    const float* Wi1 = (const float*)d_in[6];
    const float* bi1 = (const float*)d_in[7];
    const float* Wh1 = (const float*)d_in[8];
    const float* bh1 = (const float*)d_in[9];
    const float* Wfc = (const float*)d_in[10];
    const float* bfc = (const float*)d_in[11];

    rnn_scan_kernel<<<B_ / BTILE, NWAVES * 32, 0, stream>>>(
        x, hid, Wi0, bi0, Wh0, bh0, Wi1, bi1, Wh1, bh1, Wfc, bfc, (float*)d_out);
}
